// NonlinearInteractionBlock_74045236183686
// MI455X (gfx1250) — matrix operand order repaired and checked by reading, not compiled, not run on hardware
//
#include <hip/hip_runtime.h>

// ---------------------------------------------------------------------------
// MACE-style nonlinear interaction block for MI455X (gfx1250, wave32, WMMA).
//  - wconv: pre-convert all weights to bf16, zero-padded and TRANSPOSED [N][K]
//           into d_ws so LDS staging is a linear copy and B-fragments are
//           contiguous 32B chunks (2x ds_load_b128 per fragment).
//  - edge:  per-edge MLP (18->18 relu ->256) via v_wmma_f32_16x16x32_bf16,
//           tensor-product messages scatter-added with global f32 atomics
//           (msg accumulator = 41 MB -> resident in 192 MB L2).
//  - lin:   K=128 GEMM (Wls/Wlv) per node.
//  - out:   K=640 GEMM (Wss/Wsv); weights staged with
//           global_load_async_to_lds_b128 (ASYNCcnt) and the lin tile staged
//           with the Tensor Data Mover (TENSORcnt).
// ---------------------------------------------------------------------------

#define NNODES  20000
#define NEDGES  400000
#define NATTR   10
#define NEF     8
#define MLPIN   18

typedef __attribute__((ext_vector_type(16))) __bf16 v16bf;
typedef __attribute__((ext_vector_type(8)))  __bf16 v8bf;
typedef __attribute__((ext_vector_type(8)))  float  v8f;
typedef __attribute__((ext_vector_type(4)))  unsigned v4u;
typedef __attribute__((ext_vector_type(8)))  int    v8i;
typedef __attribute__((ext_vector_type(4)))  int    v4i;

// WMMA 16x16x32 bf16 fragment layouts used here (contiguous k, not the ISA table order; wave32):
//  A (16x32): lane holds row m=lane&15; elem j -> K = j + 16*(lane>=16)
//             => one contiguous 16-half chunk at K = 16*half.
//  B (32x16): lane holds col n=lane&15; elem j -> K = j + 16*(lane>=16)
//             => one contiguous 16-half chunk (weights stored [N][K] in LDS).
//  C/D:       lane holds col n=lane&15; elem j -> row m = j + 8*(lane>=16).
__device__ __forceinline__ v16bf ld_bfrag(const __bf16* p) {
  v8bf lo = *(const v8bf*)p;
  v8bf hi = *(const v8bf*)(p + 8);
  return __builtin_shufflevector(lo, hi, 0, 1, 2, 3, 4, 5, 6, 7, 8, 9, 10, 11,
                                 12, 13, 14, 15);
}
__device__ __forceinline__ v16bf ld_afrag(const __bf16* row, int half) {
  v8bf lo = *(const v8bf*)(row + 16 * half);
  v8bf hi = *(const v8bf*)(row + 16 * half + 8);
  return __builtin_shufflevector(lo, hi, 0, 1, 2, 3, 4, 5, 6, 7, 8, 9, 10, 11,
                                 12, 13, 14, 15);
}
__device__ __forceinline__ v8f wmma_bf16(v16bf a, v16bf b, v8f c) {
  return __builtin_amdgcn_wmma_f32_16x16x32_bf16(false, a, false, b, (short)0, c,
                                                 false, false);
}

// ---------------------------------------------------------------------------
__global__ void zero_kernel(float4* __restrict__ p, long long n4) {
  long long i = (long long)blockIdx.x * blockDim.x + threadIdx.x;
  long long st = (long long)gridDim.x * blockDim.x;
  float4 z = {0.f, 0.f, 0.f, 0.f};
  for (; i < n4; i += st) p[i] = z;
}

// ---------------------------------------------------------------------------
// Weight pre-pack: bf16, zero-padded, transposed to [N][K]. 107520 elements.
//   w1t [32][32], w2t [256][32], wlst/wlvt [64][128], wsst/wsvt [64][640]
// ---------------------------------------------------------------------------
__global__ __launch_bounds__(256) void wconv_kernel(
    const float* __restrict__ W1, const float* __restrict__ W2,
    const float* __restrict__ Wls, const float* __restrict__ Wlv,
    const float* __restrict__ Wss, const float* __restrict__ Wsv,
    __bf16* __restrict__ w1t, __bf16* __restrict__ w2t,
    __bf16* __restrict__ wlst, __bf16* __restrict__ wlvt,
    __bf16* __restrict__ wsst, __bf16* __restrict__ wsvt) {
  int i = blockIdx.x * 256 + threadIdx.x;
  if (i < 1024) {  // W1t
    int n = i >> 5, k = i & 31;
    w1t[i] = (__bf16)((k < MLPIN && n < MLPIN) ? W1[k * MLPIN + n] : 0.f);
  } else if (i < 9216) {  // W2t
    int j = i - 1024, n = j >> 5, k = j & 31;
    w2t[j] = (__bf16)((k < MLPIN) ? W2[k * 256 + n] : 0.f);
  } else if (i < 17408) {  // Wlst
    int j = i - 9216, n = j >> 7, k = j & 127;
    wlst[j] = (__bf16)Wls[k * 64 + n];
  } else if (i < 25600) {  // Wlvt
    int j = i - 17408, n = j >> 7, k = j & 127;
    wlvt[j] = (__bf16)Wlv[k * 64 + n];
  } else if (i < 66560) {  // Wsst
    int j = i - 25600, n = j / 640, k = j - n * 640;
    wsst[j] = (__bf16)Wss[k * 64 + n];
  } else {  // Wsvt
    int j = i - 66560, n = j / 640, k = j - n * 640;
    wsvt[j] = (__bf16)Wsv[k * 64 + n];
  }
}

// ---------------------------------------------------------------------------
// Edge kernel: 256 threads = 8 waves, 16 edges/wave, 128 edges/block.
// ---------------------------------------------------------------------------
__global__ __launch_bounds__(256) void edge_kernel(
    const float* __restrict__ node_attrs, const float* __restrict__ node_feats,
    const float* __restrict__ edge_attrs, const float* __restrict__ edge_feats,
    const int* __restrict__ edge_index, const __bf16* __restrict__ w1t,
    const __bf16* __restrict__ w2t, const float* __restrict__ b1,
    const float* __restrict__ b2, float* __restrict__ msg) {
  __shared__ alignas(16) __bf16 sW1t[32 * 32];    // [n][k] 2 KB
  __shared__ alignas(16) __bf16 sW2t[256 * 32];   // [n][k] 16 KB
  __shared__ float  sB1[32];
  __shared__ float  sB2[256];
  __shared__ alignas(16) __bf16 sX[8][16 * 32];
  __shared__ alignas(16) __bf16 sH[8][16 * 32];
  __shared__ int    sSnd[8][16];

  const int tid = threadIdx.x, lane = tid & 31, wv = tid >> 5;

  if (tid < 128) ((uint4*)sW1t)[tid] = ((const uint4*)w1t)[tid];
  for (int i = tid; i < 1024; i += 256)
    ((uint4*)sW2t)[i] = ((const uint4*)w2t)[i];
  if (tid < 32) sB1[tid] = (tid < MLPIN) ? b1[tid] : 0.f;
  sB2[tid] = b2[tid];

  const long long e0 = (long long)blockIdx.x * 128 + wv * 16;
  if (lane < 16) sSnd[wv][lane] = edge_index[e0 + lane];
  __syncthreads();

  // x = [node_attrs[snd] | edge_feats | 0-pad] as bf16 rows of 32.
  for (int i = lane; i < 16 * 32; i += 32) {
    int m = i >> 5, k = i & 31;
    long long e = e0 + m;
    float v = 0.f;
    if (k < NATTR)
      v = node_attrs[(long long)sSnd[wv][m] * NATTR + k];
    else if (k < MLPIN)
      v = edge_feats[e * NEF + (k - NATTR)];
    sX[wv][i] = (__bf16)v;
  }
  __syncthreads();

  const int m0 = lane & 15, half = lane >> 4;

  // Layer 1: h = relu(x @ W1 + b1)
  v16bf a = ld_afrag(&sX[wv][m0 * 32], half);
#pragma unroll
  for (int t = 0; t < 2; ++t) {
    v16bf b = ld_bfrag(&sW1t[(t * 16 + m0) * 32 + 16 * half]);
    v8f c = {};
    c = wmma_bf16(a, b, c);
#pragma unroll
    for (int j = 0; j < 8; ++j) {
      int row = j + 8 * half, col = t * 16 + m0;
      float v = c[j] + sB1[col];
      sH[wv][row * 32 + col] = (__bf16)(v > 0.f ? v : 0.f);
    }
  }
  __syncthreads();

  v16bf a2 = ld_afrag(&sH[wv][m0 * 32], half);

  // Layer 2 + messages. Channel ch = 16*t + n has its (w0,w1,w2,w3)
  // quadruple in N-tiles {t, t+4, t+8, t+12} of the SAME lane.
  const float rsqrt3 = 0.57735026918962576f;
  for (int t = 0; t < 4; ++t) {
    v8f acc[4];
#pragma unroll
    for (int q = 0; q < 4; ++q) {
      int col = (q * 4 + t) * 16 + m0;
      v16bf b = ld_bfrag(&sW2t[col * 32 + 16 * half]);
      v8f z = {};
      acc[q] = wmma_bf16(a2, b, z);
    }
    const int ch = t * 16 + m0;
#pragma unroll
    for (int j = 0; j < 8; ++j) {
      int m = j + 8 * half;
      long long e = e0 + m;
      int snd = sSnd[wv][m];
      int rcv = edge_index[(long long)NEDGES + e];
      float w0 = acc[0][j] + sB2[ch];
      float w1 = acc[1][j] + sB2[64 + ch];
      float w2 = acc[2][j] + sB2[128 + ch];
      float w3 = acc[3][j] + sB2[192 + ch];
      const float* nf = node_feats + (long long)snd * 256;
      float xs  = nf[ch];
      float xv0 = nf[64 + ch * 3 + 0];
      float xv1 = nf[64 + ch * 3 + 1];
      float xv2 = nf[64 + ch * 3 + 2];
      float a0  = edge_attrs[e * 4 + 0];
      float av0 = edge_attrs[e * 4 + 1];
      float av1 = edge_attrs[e * 4 + 2];
      float av2 = edge_attrs[e * 4 + 3];
      float s0 = w0 * xs * a0;
      float s1 = w3 * (xv0 * av0 + xv1 * av1 + xv2 * av2) * rsqrt3;
      float p1 = w1 * xs;  // v0 = p1 * av
      float p2 = w2 * a0;  // v1 = p2 * xv
      float* mrow = msg + (long long)rcv * 512;
      atomicAdd(mrow + ch, s0);
      atomicAdd(mrow + 64 + ch, s1);
      atomicAdd(mrow + 128 + ch * 3 + 0, p1 * av0);
      atomicAdd(mrow + 128 + ch * 3 + 1, p1 * av1);
      atomicAdd(mrow + 128 + ch * 3 + 2, p1 * av2);
      atomicAdd(mrow + 320 + ch * 3 + 0, p2 * xv0);
      atomicAdd(mrow + 320 + ch * 3 + 1, p2 * xv1);
      atomicAdd(mrow + 320 + ch * 3 + 2, p2 * xv2);
    }
  }
}

// ---------------------------------------------------------------------------
// Linear kernel: lin_s = ms@Wls*scale, lin_v = mv@Wlv*scale (K=128 GEMM).
// 64 threads = 2 waves, 16 nodes/wave (20000/32 = 625 blocks exact).
// ---------------------------------------------------------------------------
__global__ __launch_bounds__(64) void lin_kernel(
    const float* __restrict__ msg, const __bf16* __restrict__ wlst,
    const __bf16* __restrict__ wlvt, float* __restrict__ lin) {
  __shared__ alignas(16) __bf16 sWls[64 * 128];  // [n][k]
  __shared__ alignas(16) __bf16 sWlv[64 * 128];  // [n][k]
  const int tid = threadIdx.x, lane = tid & 31, wv = tid >> 5;
  for (int i = tid; i < 1024; i += 64) {
    ((uint4*)sWls)[i] = ((const uint4*)wlst)[i];
    ((uint4*)sWlv)[i] = ((const uint4*)wlvt)[i];
  }
  __syncthreads();

  const float scale = 0.00441941738241592f;  // 1/(sqrt(128)*20)
  const int n0 = blockIdx.x * 32 + wv * 16;
  const int m0 = lane & 15, half = lane >> 4;

  for (int chan = 0; chan < 4; ++chan) {  // 0: scalar, 1..3: vector c
    const __bf16* sW = (chan == 0) ? sWls : sWlv;
    v8f acc[4] = {};
    for (int ks = 0; ks < 4; ++ks) {
      v16bf a;
      const float* mrow = msg + (long long)(n0 + m0) * 512;
#pragma unroll
      for (int j = 0; j < 16; ++j) {
        int K = j + 16 * half + ks * 32;
        float v = (chan == 0) ? mrow[K] : mrow[128 + K * 3 + (chan - 1)];
        a[j] = (__bf16)v;
      }
#pragma unroll
      for (int nt = 0; nt < 4; ++nt) {
        v16bf b = ld_bfrag(&sW[(nt * 16 + m0) * 128 + 16 * half + ks * 32]);
        acc[nt] = wmma_bf16(a, b, acc[nt]);
      }
    }
#pragma unroll
    for (int nt = 0; nt < 4; ++nt) {
#pragma unroll
      for (int j = 0; j < 8; ++j) {
        int row = j + 8 * half, col = nt * 16 + m0;
        float v = acc[nt][j] * scale;
        float* lrow = lin + (long long)(n0 + row) * 256;
        if (chan == 0)
          lrow[col] = v;
        else
          lrow[64 + col * 3 + (chan - 1)] = v;
      }
    }
  }
}

// ---------------------------------------------------------------------------
// Output kernel: out = (lin (x) attrs) @ W * sk (K=640 GEMM).
// Weight tile (80 KB bf16) staged via GLOBAL_LOAD_ASYNC_TO_LDS_B128;
// lin tile (32 KB f32) staged via the Tensor Data Mover.
// Launched twice: vec=0 (Wss -> out_s), vec=1 (Wsv -> out_v).
// ---------------------------------------------------------------------------
__global__ __launch_bounds__(64) void out_kernel(
    const float* __restrict__ node_attrs, const float* __restrict__ lin,
    const __bf16* __restrict__ Wt, float* __restrict__ out, int vec) {
  __shared__ alignas(16) __bf16 sW[64 * 640];    // [n][k] 80 KB
  __shared__ alignas(16) float  sLin[32 * 256];  // 32 KB
  __shared__ float sAttr[32 * NATTR];
  const int tid = threadIdx.x, lane = tid & 31, wv = tid >> 5;
  const int nb = blockIdx.x * 32;

  // 80 KB weight tile: 80 iters x 64 lanes x 16 B, ASYNCcnt-tracked.
  {
    unsigned lds_base = (unsigned)(size_t)&sW[0];
    unsigned long long gbase = (unsigned long long)Wt;
    for (int it = 0; it < 80; ++it) {
      unsigned off = (unsigned)(it * 64 + tid) * 16u;
      unsigned lds_addr = lds_base + off;
      asm volatile("global_load_async_to_lds_b128 %0, %1, %2"
                   :
                   : "v"(lds_addr), "v"(off), "s"(gbase)
                   : "memory");
    }
  }
  // 32 KB lin tile via TDM (one wave issues; EXEC-independent DMA).
  if (wv == 0) {
    unsigned lds_off = (unsigned)(size_t)&sLin[0];
    unsigned long long ga =
        (unsigned long long)(const void*)(lin + (long long)nb * 256);
    v4u g0;
    g0[0] = 1u;  // count=1, user descriptor, no gather
    g0[1] = lds_off;
    g0[2] = (unsigned)(ga & 0xffffffffu);
    g0[3] = (unsigned)((ga >> 32) & 0x01ffffffu) | (2u << 30);  // type=2
    v8i g1;
    g1[0] = (int)(2u << 16);      // data_size = 4 B
    g1[1] = (int)(8192u << 16);   // tensor_dim0[15:0]
    g1[2] = (int)(1u << 16);      // tensor_dim0[31:16]=0 | tensor_dim1=1
    g1[3] = (int)(8192u << 16);   // tile_dim0 = 8192
    g1[4] = 1;                    // tile_dim1 = 1, tile_dim2 = 0
    g1[5] = 8192;                 // tensor_dim0_stride lo
    g1[6] = 0;
    g1[7] = 0;
    v4i z4 = {0, 0, 0, 0};
    v8i z8 = {0, 0, 0, 0, 0, 0, 0, 0};
    __builtin_amdgcn_tensor_load_to_lds(g0, g1, z4, z4, z8, 0);
  }
  for (int i = tid; i < 32 * NATTR; i += 64)
    sAttr[i] = node_attrs[(long long)nb * NATTR + i];

  asm volatile("s_wait_asynccnt 0" ::: "memory");
  __builtin_amdgcn_s_wait_tensorcnt(0);
  __syncthreads();

  const float sk = 0.03952847075210474f;  // 1/sqrt(64*10)
  const int m0 = lane & 15, half = lane >> 4;
  const int ml = wv * 16 + m0;  // A-row node (local)
  const int nchan = vec ? 3 : 1;

  for (int cc = 0; cc < nchan; ++cc) {
    v8f acc[4] = {};
    for (int ks = 0; ks < 20; ++ks) {
      v16bf a;
#pragma unroll
      for (int j = 0; j < 16; ++j) {
        int K = j + 16 * half + ks * 32;  // K in [0,640)
        int i = K / NATTR, aa = K - i * NATTR;
        float lv = vec ? sLin[ml * 256 + 64 + i * 3 + cc] : sLin[ml * 256 + i];
        a[j] = (__bf16)(lv * sAttr[ml * NATTR + aa]);
      }
#pragma unroll
      for (int nt = 0; nt < 4; ++nt) {
        v16bf b = ld_bfrag(&sW[(nt * 16 + m0) * 640 + 16 * half + ks * 32]);
        acc[nt] = wmma_bf16(a, b, acc[nt]);
      }
    }
#pragma unroll
    for (int nt = 0; nt < 4; ++nt) {
#pragma unroll
      for (int j = 0; j < 8; ++j) {
        int row = wv * 16 + j + 8 * half, col = nt * 16 + m0;
        float v = acc[nt][j] * sk;
        float* orow = out + (long long)(nb + row) * 256;
        if (vec)
          orow[64 + col * 3 + cc] = v;
        else
          orow[col] = v;
      }
    }
  }
}

// ---------------------------------------------------------------------------
extern "C" void kernel_launch(void* const* d_in, const int* in_sizes, int n_in,
                              void* d_out, int out_size, void* d_ws,
                              size_t ws_size, hipStream_t stream) {
  const float* node_attrs = (const float*)d_in[0];
  const float* node_feats = (const float*)d_in[1];
  const float* edge_attrs = (const float*)d_in[2];
  const float* edge_feats = (const float*)d_in[3];
  const int*   edge_index = (const int*)d_in[4];  // harness passes indices as i32
  const float* W1  = (const float*)d_in[5];
  const float* b1  = (const float*)d_in[6];
  const float* W2  = (const float*)d_in[7];
  const float* b2  = (const float*)d_in[8];
  const float* Wls = (const float*)d_in[9];
  const float* Wlv = (const float*)d_in[10];
  const float* Wss = (const float*)d_in[11];
  const float* Wsv = (const float*)d_in[12];
  float* out = (float*)d_out;

  float*  msg = (float*)d_ws;                    // NNODES*512 f32 (40.96 MB)
  float*  lin = msg + (long long)NNODES * 512;   // NNODES*256 f32 (20.48 MB)
  __bf16* wb  = (__bf16*)(lin + (long long)NNODES * 256);
  __bf16* w1t  = wb;            // 1024
  __bf16* w2t  = w1t + 1024;    // 8192
  __bf16* wlst = w2t + 8192;    // 8192
  __bf16* wlvt = wlst + 8192;   // 8192
  __bf16* wsst = wlvt + 8192;   // 40960
  __bf16* wsvt = wsst + 40960;  // 40960

  zero_kernel<<<2048, 256, 0, stream>>>((float4*)msg,
                                        (long long)NNODES * 512 / 4);
  wconv_kernel<<<420, 256, 0, stream>>>(W1, W2, Wls, Wlv, Wss, Wsv, w1t, w2t,
                                        wlst, wlvt, wsst, wsvt);
  edge_kernel<<<NEDGES / 128, 256, 0, stream>>>(node_attrs, node_feats,
                                                edge_attrs, edge_feats,
                                                edge_index, w1t, w2t, b1, b2,
                                                msg);
  lin_kernel<<<NNODES / 32, 64, 0, stream>>>(msg, wlst, wlvt, lin);
  out_kernel<<<NNODES / 32, 64, 0, stream>>>(node_attrs, lin, wsst, out, 0);
  out_kernel<<<NNODES / 32, 64, 0, stream>>>(node_attrs, lin, wsvt, out, 1);
}
